// MaskGeneratingModel_57054345560499
// MI455X (gfx1250) — matrix operand order repaired and checked by reading, not compiled, not run on hardware
//
#include <hip/hip_runtime.h>
#include <hip/hip_bf16.h>

typedef __bf16 bf16_t;
typedef __bf16 v16bf __attribute__((ext_vector_type(16)));
typedef __bf16 v8bf  __attribute__((ext_vector_type(8)));
typedef float  v8f   __attribute__((ext_vector_type(8)));

#define N_B 2
#define L_S 64
#define H_D 512
#define V_S 8192

// ---------------- kernel 0: W_lm f32 -> bf16 ----------------
__global__ void k_cvt(const float* __restrict__ src, bf16_t* __restrict__ dst, int n) {
    int i = blockIdx.x * blockDim.x + threadIdx.x;
    int stride = gridDim.x * blockDim.x;
    for (; i < n; i += stride) dst[i] = (bf16_t)src[i];
}

// ---------------- kernel 1: per-row MLPs (policy h + value scalar) ----------------
__global__ __launch_bounds__(256)
void k_mlp(const float* __restrict__ x,            // [128,512]
           const float* __restrict__ Wp1, const float* __restrict__ bp1,
           const float* __restrict__ Wp2, const float* __restrict__ bp2,
           const float* __restrict__ Wv1, const float* __restrict__ bv1,
           const float* __restrict__ Wv2, const float* __restrict__ bv2,
           const float* __restrict__ Wv3, const float* __restrict__ bv3,
           float* __restrict__ h_out,              // [128,512]
           float* __restrict__ vscal)              // [128]
{
    __shared__ float xs[H_D];
    __shared__ float ts[H_D];
    __shared__ float red[256];
    const int row = blockIdx.x;
    const int tid = threadIdx.x;

    for (int k = tid; k < H_D; k += 256) xs[k] = x[(size_t)row * H_D + k];
    __syncthreads();

    // policy layer 1: ReLU(x @ Wp1^T + bp1)
    for (int o = tid; o < H_D; o += 256) {
        float acc = bp1[o];
        const float* w = Wp1 + (size_t)o * H_D;
        for (int k = 0; k < H_D; ++k) acc = fmaf(xs[k], w[k], acc);
        ts[o] = fmaxf(acc, 0.f);
    }
    __syncthreads();
    // policy layer 2
    for (int o = tid; o < H_D; o += 256) {
        float acc = bp2[o];
        const float* w = Wp2 + (size_t)o * H_D;
        for (int k = 0; k < H_D; ++k) acc = fmaf(ts[k], w[k], acc);
        h_out[(size_t)row * H_D + o] = acc;
    }
    __syncthreads();
    // value layer 1 (reuse ts)
    for (int o = tid; o < H_D; o += 256) {
        float acc = bv1[o];
        const float* w = Wv1 + (size_t)o * H_D;
        for (int k = 0; k < H_D; ++k) acc = fmaf(xs[k], w[k], acc);
        ts[o] = fmaxf(acc, 0.f);
    }
    __syncthreads();
    // value layer 2 fused with Wv3 dot
    float local = 0.f;
    for (int o = tid; o < H_D; o += 256) {
        float acc = bv2[o];
        const float* w = Wv2 + (size_t)o * H_D;
        for (int k = 0; k < H_D; ++k) acc = fmaf(ts[k], w[k], acc);
        local = fmaf(acc, Wv3[o], local);
    }
    red[tid] = local;
    __syncthreads();
    if (tid == 0) {
        float s = bv3[0];
        for (int t = 0; t < 256; ++t) s += red[t];
        vscal[row] = s;
    }
}

// ---------------- kernel 2: fused pairwise-logit GEMM + softmax-at-label ----------------
// One block per (n,i). 8 waves sweep the vocab in DOUBLE tiles (32 cols):
// each A tile loaded from LDS feeds two WMMAs (two B operands), halving DS
// traffic per v_wmma and giving two independent accumulation chains.
__global__ __launch_bounds__(256)
void k_policy(const float*  __restrict__ h_g,     // [128,512] policy-MLP output
              const bf16_t* __restrict__ wlm,     // [8192,512] bf16
              const float*  __restrict__ am,      // [2,64] attention mask
              const int*    __restrict__ labels,  // [2,64]
              float* __restrict__ out)            // policy probs at out[0..127]
{
    extern __shared__ char smem[];
    bf16_t* pair   = (bf16_t*)smem;                    // 64*512 bf16 = 64KB
    float*  sumex  = (float*)(smem + 65536);           // [64]
    float*  llog   = (float*)(smem + 65536 + 256);     // [64]
    int*    safe_s = (int*)  (smem + 65536 + 512);     // [64]
    float*  maskf  = (float*)(smem + 65536 + 768);     // [64]
    float*  red1   = (float*)(smem + 65536 + 1024);    // [64]
    float*  red2   = (float*)(smem + 65536 + 1280);    // [64]

    const int blk  = blockIdx.x;      // 0..127
    const int n    = blk >> 6;
    const int ipos = blk & 63;
    const int tid  = threadIdx.x;
    const int lane = tid & 31;
    const int wave = tid >> 5;        // 0..7
    const int m    = lane & 15;       // A-row within 16-tile / D-col
    const int kg   = lane >> 4;       // K-group (also D row-group mg)
    const int ncol = lane & 15;

    // stage shifted labels + masks, init reductions
    if (tid < L_S) {
        int j   = tid;
        int lab = (j < L_S - 1) ? labels[n * L_S + j + 1] : -100;
        safe_s[j] = lab > 0 ? lab : 0;
        maskf[j]  = (lab == -100) ? 1.f : 0.f;
        sumex[j]  = 0.f;
        llog[j]   = 0.f;
    }
    // pair[j,h] = h[n,i,h] * h[n,j,h]  (bf16 in LDS)
    const float* hi = h_g + ((size_t)(n * L_S + ipos)) * H_D;
    for (int idx = tid; idx < L_S * H_D; idx += 256) {
        int j  = idx >> 9;            // / 512
        int hh = idx & (H_D - 1);
        float pv = hi[hh] * h_g[((size_t)(n * L_S + j)) * H_D + hh];
        pair[idx] = (bf16_t)pv;
    }
    __syncthreads();

    // this lane's 32 fixed output rows: j = mt*16 + 8*kg + r
    int safe_r[4][8];
    #pragma unroll
    for (int mt = 0; mt < 4; ++mt)
        #pragma unroll
        for (int r = 0; r < 8; ++r)
            safe_r[mt][r] = safe_s[mt * 16 + (kg << 3) + r];

    float partial[4][8];
    #pragma unroll
    for (int mt = 0; mt < 4; ++mt)
        #pragma unroll
        for (int r = 0; r < 8; ++r) partial[mt][r] = 0.f;

    const v8f vzero = {0.f, 0.f, 0.f, 0.f, 0.f, 0.f, 0.f, 0.f};

    // sweep vocab in double tiles (32 columns); wave-uniform -> EXEC all-1s
    for (int vt = wave; vt < V_S / 32; vt += 8) {
        const int vbase = vt * 32;
        v8f acc0[4] = {vzero, vzero, vzero, vzero};
        v8f acc1[4] = {vzero, vzero, vzero, vzero};

        for (int kb = 0; kb < H_D; kb += 32) {
            // B-matrix 32x16: lane (16*kgrp + col) holds 16 contiguous K along H
            const bf16_t* wb = wlm + (size_t)(vbase + ncol) * H_D + kb + kg * 16;
            v16bf b0 = *(const v16bf*)(wb);
            v16bf b1 = *(const v16bf*)(wb + 16 * H_D);
            #pragma unroll
            for (int mt = 0; mt < 4; ++mt) {
                // A-matrix 16x32: lane row m, 16 contiguous K=[kb+16kg,+16)
                const bf16_t* pp = pair + (size_t)(mt * 16 + m) * H_D + kb + kg * 16;
                v8bf a0 = *(const v8bf*)(pp);
                v8bf a1 = *(const v8bf*)(pp + 8);
                v16bf a = __builtin_shufflevector(a0, a1,
                          0,1,2,3,4,5,6,7,8,9,10,11,12,13,14,15);
                acc0[mt] = __builtin_amdgcn_wmma_f32_16x16x32_bf16(
                               false, a, false, b0, (short)0, acc0[mt], false, false);
                acc1[mt] = __builtin_amdgcn_wmma_f32_16x16x32_bf16(
                               false, a, false, b1, (short)0, acc1[mt], false, false);
            }
        }
        // consume logits: running sum-of-exp + capture logit at label column
        #pragma unroll
        for (int mt = 0; mt < 4; ++mt) {
            #pragma unroll
            for (int r = 0; r < 8; ++r) {
                float lg0 = acc0[mt][r];
                float lg1 = acc1[mt][r];
                partial[mt][r] += __expf(lg0) + __expf(lg1);
                int jr = mt * 16 + (kg << 3) + r;
                if (vbase + ncol == safe_r[mt][r])      llog[jr] = lg0;
                if (vbase + 16 + ncol == safe_r[mt][r]) llog[jr] = lg1;
            }
        }
    }

    // reduce per-row sum-of-exp into LDS
    #pragma unroll
    for (int mt = 0; mt < 4; ++mt)
        #pragma unroll
        for (int r = 0; r < 8; ++r)
            atomicAdd(&sumex[mt * 16 + (kg << 3) + r], partial[mt][r]);
    __syncthreads();

    // lm_probs row -> causal-masked average
    if (tid < L_S) {
        int j    = tid;
        float p  = (maskf[j] > 0.5f) ? 0.f : __expf(llog[j]) / sumex[j];
        float cz = (j >= ipos) ? am[n * L_S + ipos] * am[n * L_S + j] : 0.f;
        red1[j] = p * cz;
        red2[j] = cz;
    }
    __syncthreads();
    if (tid == 0) {
        float s1 = 0.f, s2 = 0.f;
        for (int j = 0; j < L_S; ++j) { s1 += red1[j]; s2 += red2[j]; }
        out[n * L_S + ipos] = s1 / (s2 + 1.f);
    }
}

// ---------------- kernel 3: masked value mean ----------------
__global__ void k_value(const float* __restrict__ vscal,
                        const float* __restrict__ rm,
                        float* __restrict__ out)
{
    int n = threadIdx.x;
    if (n < N_B) {
        float s = 0.f, sr = 0.f;
        for (int l = 0; l < L_S; ++l) {
            s  += vscal[n * L_S + l] * rm[n * L_S + l];
            sr += rm[n * L_S + l];
        }
        out[N_B * L_S + n] = s / (sr + 1.f);   // value at out[128..129]
    }
}

extern "C" void kernel_launch(void* const* d_in, const int* in_sizes, int n_in,
                              void* d_out, int out_size, void* d_ws, size_t ws_size,
                              hipStream_t stream) {
    const float* hs  = (const float*)d_in[0];
    const float* am  = (const float*)d_in[1];
    const float* rm  = (const float*)d_in[2];
    const int*   lab = (const int*)  d_in[3];
    const float* Wlm = (const float*)d_in[4];
    const float* Wp1 = (const float*)d_in[5];
    const float* bp1 = (const float*)d_in[6];
    const float* Wp2 = (const float*)d_in[7];
    const float* bp2 = (const float*)d_in[8];
    const float* Wv1 = (const float*)d_in[9];
    const float* bv1 = (const float*)d_in[10];
    const float* Wv2 = (const float*)d_in[11];
    const float* bv2 = (const float*)d_in[12];
    const float* Wv3 = (const float*)d_in[13];
    const float* bv3 = (const float*)d_in[14];
    float* out = (float*)d_out;

    char* ws = (char*)d_ws;
    bf16_t* wlm_bf = (bf16_t*)ws;                                    // 8 MB
    float*  h_g    = (float*)(ws + (size_t)V_S * H_D * 2);           // 256 KB
    float*  vscal  = (float*)(ws + (size_t)V_S * H_D * 2
                                 + (size_t)N_B * L_S * H_D * 4);     // 512 B

    k_cvt<<<512, 256, 0, stream>>>(Wlm, wlm_bf, V_S * H_D);
    k_mlp<<<N_B * L_S, 256, 0, stream>>>(hs, Wp1, bp1, Wp2, bp2,
                                         Wv1, bv1, Wv2, bv2, Wv3, bv3,
                                         h_g, vscal);
    size_t smem = 65536 + 6 * 256;   // pair + small reduction arrays
    k_policy<<<N_B * L_S, 256, smem, stream>>>(h_g, wlm_bf, am, lab, out);
    k_value<<<1, 64, 0, stream>>>(vscal, rm, out);
}
